// GraphCare_79860621902251
// MI455X (gfx1250) — matrix operand order repaired and checked by reading, not compiled, not run on hardware
//
#include <hip/hip_runtime.h>
#include <hip/hip_bf16.h>
#include <cmath>

#define DEVFN __device__ __forceinline__

typedef __bf16 hbf;
typedef hbf   v8bf  __attribute__((ext_vector_type(8)));
typedef hbf   v16bf __attribute__((ext_vector_type(16)));
typedef float v8f   __attribute__((ext_vector_type(8)));

namespace gc {
constexpr int B = 64, V = 20, NN1 = 2001, NE1 = 501;
constexpr int N = 32768, E = 131072;
constexpr int D = 768, H = 256, OUT = 25, L = 2;
constexpr float DECAY = 0.01f;
// padded GEMM geometry (M -> mult of 128, N -> mult of 64, K -> mult of 32)
constexpr int MP_EMB = 2048;   // node_emb rows 2001 -> 2048
constexpr int MP_VIS = 1280;   // B*V already mult of 128
constexpr int NP_NN  = 2048;   // alpha_w rows 2001 -> 2048
constexpr int KP_NN  = 2016;   // K = 2001 -> 2016
}

// ---------------------------------------------------------------------------
// f32 -> bf16 convert with zero padding; optional fused add of second source.
// dst is [Rp, Kp] row-major bf16; src is [R, K] row-major f32.
// ---------------------------------------------------------------------------
template <bool ADD2>
__global__ void cvt_bf16_kernel(const float* __restrict__ src,
                                const float* __restrict__ src2,
                                hbf* __restrict__ dst,
                                int R, int K, int Kp, int total) {
  int idx = blockIdx.x * blockDim.x + threadIdx.x;
  if (idx >= total) return;
  int r = idx / Kp, k = idx - r * Kp;
  float v = 0.0f;
  if (r < R && k < K) {
    v = src[(size_t)r * K + k];
    if (ADD2) v += src2[(size_t)r * K + k];
  }
  dst[idx] = (hbf)v;
}

// ---------------------------------------------------------------------------
// C[m,n] = act( sum_k A[m,k] * W[n,k] + bias[n] ),  A/W pre-padded bf16.
// Block = 256 threads = 8 waves (4Mx2N); wave computes 32x32 via 2x2 WMMAs.
// Block tile = 128(M) x 64(N). No guards in the K loop: operands are padded.
// ---------------------------------------------------------------------------
template <bool RELU>
__global__ __launch_bounds__(256) void wmma_gemm_bf16(
    const hbf* __restrict__ A, const hbf* __restrict__ W,
    const float* __restrict__ bias, float* __restrict__ C,
    int M, int Nn, int Kp) {
  const int lane = threadIdx.x & 31;
  const int wv   = threadIdx.x >> 5;
  const int m0   = blockIdx.x * 128 + (wv >> 1) * 32;
  const int n0   = blockIdx.y * 64 + (wv & 1) * 32;
  const int half = lane >> 4;

  const hbf* __restrict__ arow0 = A + (size_t)(m0 + (lane & 15)) * Kp;
  const hbf* __restrict__ arow1 = arow0 + (size_t)16 * Kp;
  const hbf* __restrict__ wrow0 = W + (size_t)(n0 + (lane & 15)) * Kp;
  const hbf* __restrict__ wrow1 = wrow0 + (size_t)16 * Kp;

  v8f acc00 = {}, acc01 = {}, acc10 = {}, acc11 = {};
  for (int kk = 0; kk < Kp; kk += 32) {
    __builtin_prefetch(arow0 + kk + 256, 0, 1);  // global_prefetch_b8
    __builtin_prefetch(wrow0 + kk + 256, 0, 1);
    // A fragment: lane holds row m, K runs [kk+8h, +8) and [kk+16+8h, +8)
    const int ka = kk + 8 * half;
    v8bf a0lo = *(const v8bf*)(arow0 + ka);
    v8bf a0hi = *(const v8bf*)(arow0 + ka + 16);
    v8bf a1lo = *(const v8bf*)(arow1 + ka);
    v8bf a1hi = *(const v8bf*)(arow1 + ka + 16);
    v16bf a0 = __builtin_shufflevector(a0lo, a0hi, 0, 1, 2, 3, 4, 5, 6, 7,
                                       8, 9, 10, 11, 12, 13, 14, 15);
    v16bf a1 = __builtin_shufflevector(a1lo, a1hi, 0, 1, 2, 3, 4, 5, 6, 7,
                                       8, 9, 10, 11, 12, 13, 14, 15);
    // B fragment: lane holds row n of W, K runs [kk+8h, +8) and [kk+16+8h, +8)
    const int kb = kk + 8 * half;
    v8bf b0lo = *(const v8bf*)(wrow0 + kb);
    v8bf b0hi = *(const v8bf*)(wrow0 + kb + 16);
    v8bf b1lo = *(const v8bf*)(wrow1 + kb);
    v8bf b1hi = *(const v8bf*)(wrow1 + kb + 16);
    v16bf b0 = __builtin_shufflevector(b0lo, b0hi, 0, 1, 2, 3, 4, 5, 6, 7,
                                       8, 9, 10, 11, 12, 13, 14, 15);
    v16bf b1 = __builtin_shufflevector(b1lo, b1hi, 0, 1, 2, 3, 4, 5, 6, 7,
                                       8, 9, 10, 11, 12, 13, 14, 15);

    acc00 = __builtin_amdgcn_wmma_f32_16x16x32_bf16(false, a0, false, b0, (short)0, acc00, false, false);
    acc01 = __builtin_amdgcn_wmma_f32_16x16x32_bf16(false, a0, false, b1, (short)0, acc01, false, false);
    acc10 = __builtin_amdgcn_wmma_f32_16x16x32_bf16(false, a1, false, b0, (short)0, acc10, false, false);
    acc11 = __builtin_amdgcn_wmma_f32_16x16x32_bf16(false, a1, false, b1, (short)0, acc11, false, false);
  }

  // C/D tile: lane&15 = column, VGPR r = row r + 8*(lane>>4)
  const int nA = n0 + (lane & 15);
  const int nB = nA + 16;
  const int mb0 = m0 + (half << 3);
  const int mb1 = mb0 + 16;
#pragma unroll
  for (int r = 0; r < 8; ++r) {
    const int ma = mb0 + r, mc = mb1 + r;
    if (nA < Nn) {
      const float bv = bias[nA];
      if (ma < M) {
        float v = acc00[r] + bv;
        if (RELU) v = fmaxf(v, 0.f);
        C[(size_t)ma * Nn + nA] = v;
      }
      if (mc < M) {
        float v = acc10[r] + bv;
        if (RELU) v = fmaxf(v, 0.f);
        C[(size_t)mc * Nn + nA] = v;
      }
    }
    if (nB < Nn) {
      const float bv = bias[nB];
      if (ma < M) {
        float v = acc01[r] + bv;
        if (RELU) v = fmaxf(v, 0.f);
        C[(size_t)ma * Nn + nB] = v;
      }
      if (mc < M) {
        float v = acc11[r] + bv;
        if (RELU) v = fmaxf(v, 0.f);
        C[(size_t)mc * Nn + nB] = v;
      }
    }
  }
}

// ---------------------------------------------------------------------------
// Small supporting kernels
// ---------------------------------------------------------------------------
__global__ void zero_kernel(float4* __restrict__ p, int n4) {
  int i = blockIdx.x * blockDim.x + threadIdx.x;
  if (i < n4) p[i] = make_float4(0.f, 0.f, 0.f, 0.f);
}

__global__ __launch_bounds__(64) void gather_rows_kernel(
    const int* __restrict__ ids, const float* __restrict__ table,
    float* __restrict__ out) {
  const int row = blockIdx.x;
  const float4* src = (const float4*)(table + (size_t)ids[row] * gc::H);
  float4* dst = (float4*)(out + (size_t)row * gc::H);
  dst[threadIdx.x] = src[threadIdx.x];
}

// u[l,d] = sum_h wr_w[l,h] * lin_w[h,d]
__global__ void fold_wr_kernel(const float* __restrict__ wr_w,
                               const float* __restrict__ lin_w,
                               float* __restrict__ u) {
  int t = blockIdx.x * blockDim.x + threadIdx.x;
  if (t >= gc::L * gc::D) return;
  int l = t / gc::D, d = t % gc::D;
  float acc = 0.f;
  for (int h = 0; h < gc::H; ++h) acc += wr_w[l * gc::H + h] * lin_w[h * gc::D + d];
  u[t] = acc;
}

// wrel[l,e] = dot(edge_emb_w[e,:], u[l,:]) + sum_h lin_b[h]*wr_w[l,h] + wr_b[l]
__global__ void wrel_kernel(const float* __restrict__ edge_emb_w,
                            const float* __restrict__ u,
                            const float* __restrict__ lin_b,
                            const float* __restrict__ wr_w,
                            const float* __restrict__ wr_b,
                            float* __restrict__ wrel) {
  int t = blockIdx.x * blockDim.x + threadIdx.x;
  if (t >= gc::L * gc::NE1) return;
  int l = t / gc::NE1, e = t % gc::NE1;
  float acc = 0.f;
  for (int d = 0; d < gc::D; ++d) acc += edge_emb_w[(size_t)e * gc::D + d] * u[l * gc::D + d];
  float cb = 0.f;
  for (int h = 0; h < gc::H; ++h) cb += lin_b[h] * wr_w[l * gc::H + h];
  wrel[t] = acc + cb + wr_b[l];
}

// beta[b,v] = tanh(dot(visit[b,v,:], beta_w[l]) + beta_b[l]) * exp(DECAY*(V-v))
__global__ void beta_kernel(const float* __restrict__ visit,
                            const float* __restrict__ bw,
                            const float* __restrict__ bb,
                            float* __restrict__ beta) {
  int t = blockIdx.x * blockDim.x + threadIdx.x;
  if (t >= gc::B * gc::V) return;
  int v = t % gc::V;
  const float* row = visit + (size_t)t * gc::NN1;
  float acc = 0.f;
  for (int n = 0; n < gc::NN1; ++n) acc += row[n] * bw[n];
  beta[t] = tanhf(acc + bb[0]) * __expf(gc::DECAY * (float)(gc::V - v));
}

// attn[b,m] = sum_v softmax_v(h[b,:,m]) * beta[b,v]
__global__ void softmax_attn_kernel(const float* __restrict__ h,
                                    const float* __restrict__ beta,
                                    float* __restrict__ attn) {
  int t = blockIdx.x * blockDim.x + threadIdx.x;
  if (t >= gc::B * gc::NN1) return;
  int b = t / gc::NN1, m = t - b * gc::NN1;
  float hv[gc::V];
  float mx = -1e30f;
#pragma unroll
  for (int v = 0; v < gc::V; ++v) {
    hv[v] = h[(size_t)(b * gc::V + v) * gc::NN1 + m];
    mx = fmaxf(mx, hv[v]);
  }
  float se = 0.f, sw = 0.f;
#pragma unroll
  for (int v = 0; v < gc::V; ++v) {
    float ex = __expf(hv[v] - mx);
    se += ex;
    sw += ex * beta[b * gc::V + v];
  }
  attn[t] = sw / se;
}

// per edge: agg[dst,:] += relu(x[src,:] * attn[batch[src], node_ids[src]] * wrel[edge_id])
__global__ __launch_bounds__(64) void edge_msg_kernel(
    const int* __restrict__ edge_index, const int* __restrict__ edge_ids,
    const int* __restrict__ node_ids, const int* __restrict__ batch,
    const float* __restrict__ attn, const float* __restrict__ wrel_l,
    const float* __restrict__ x, float* __restrict__ agg) {
  const int e = blockIdx.x;
  const int s = edge_index[e];
  const int d = edge_index[gc::E + e];
  const float coef = attn[(size_t)batch[s] * gc::NN1 + node_ids[s]] * wrel_l[edge_ids[e]];
  const float4 xv = ((const float4*)(x + (size_t)s * gc::H))[threadIdx.x];
  float* a = agg + (size_t)d * gc::H + threadIdx.x * 4;
  const float m0 = fmaxf(xv.x * coef, 0.f);
  const float m1 = fmaxf(xv.y * coef, 0.f);
  const float m2 = fmaxf(xv.z * coef, 0.f);
  const float m3 = fmaxf(xv.w * coef, 0.f);
  __hip_atomic_fetch_add(a + 0, m0, __ATOMIC_RELAXED, __HIP_MEMORY_SCOPE_AGENT);
  __hip_atomic_fetch_add(a + 1, m1, __ATOMIC_RELAXED, __HIP_MEMORY_SCOPE_AGENT);
  __hip_atomic_fetch_add(a + 2, m2, __ATOMIC_RELAXED, __HIP_MEMORY_SCOPE_AGENT);
  __hip_atomic_fetch_add(a + 3, m3, __ATOMIC_RELAXED, __HIP_MEMORY_SCOPE_AGENT);
}

__global__ __launch_bounds__(256) void pool_kernel(const float* __restrict__ x,
                                                   float* __restrict__ xg) {
  const int b = blockIdx.x, h = threadIdx.x;
  const int per = gc::N / gc::B;  // 512 (batch is sorted, contiguous)
  float s = 0.f;
  for (int j = 0; j < per; ++j) s += x[(size_t)(b * per + j) * gc::H + h];
  xg[b * gc::H + h] = s * (1.0f / (float)per);
}

__global__ void denom_kernel(const float* __restrict__ ehr, float* __restrict__ denom) {
  int b = threadIdx.x;
  if (b >= gc::B) return;
  float s = 0.f;
  for (int m = 0; m < gc::NN1; ++m) s += ehr[(size_t)b * gc::NN1 + m];
  denom[b] = fmaxf(s, 1.0f);
}

__global__ void tnode_kernel(const float* __restrict__ ehr,
                             const float* __restrict__ emb,
                             const float* __restrict__ denom,
                             float* __restrict__ t) {
  int idx = blockIdx.x * blockDim.x + threadIdx.x;
  if (idx >= gc::B * gc::D) return;
  int b = idx / gc::D, d = idx - b * gc::D;
  float acc = 0.f;
  for (int m = 0; m < gc::NN1; ++m) acc += ehr[(size_t)b * gc::NN1 + m] * emb[(size_t)m * gc::D + d];
  t[idx] = acc / denom[b];
}

__global__ void xnode_kernel(const float* __restrict__ t,
                             const float* __restrict__ lin_w,
                             const float* __restrict__ lin_b,
                             float* __restrict__ xn) {
  int idx = blockIdx.x * blockDim.x + threadIdx.x;
  if (idx >= gc::B * gc::H) return;
  int b = idx / gc::H, h = idx - b * gc::H;
  float acc = lin_b[h];
  for (int d = 0; d < gc::D; ++d) acc += t[(size_t)b * gc::D + d] * lin_w[(size_t)h * gc::D + d];
  xn[idx] = acc;
}

__global__ void mlp_kernel(const float* __restrict__ xg, const float* __restrict__ xn,
                           const float* __restrict__ mlp_w, const float* __restrict__ mlp_b,
                           float* __restrict__ out) {
  int idx = blockIdx.x * blockDim.x + threadIdx.x;
  if (idx >= gc::B * gc::OUT) return;
  int b = idx / gc::OUT, o = idx - b * gc::OUT;
  float acc = mlp_b[o];
  for (int h = 0; h < gc::H; ++h) {
    acc += xg[b * gc::H + h] * mlp_w[(size_t)o * (2 * gc::H) + h];
    acc += xn[b * gc::H + h] * mlp_w[(size_t)o * (2 * gc::H) + gc::H + h];
  }
  out[idx] = acc;
}

// ---------------------------------------------------------------------------
extern "C" void kernel_launch(void* const* d_in, const int* in_sizes, int n_in,
                              void* d_out, int out_size, void* d_ws, size_t ws_size,
                              hipStream_t stream) {
  using namespace gc;
  (void)in_sizes; (void)n_in; (void)out_size; (void)ws_size;

  const int*   node_ids   = (const int*)d_in[0];
  const int*   edge_ids   = (const int*)d_in[1];
  const int*   edge_index = (const int*)d_in[2];
  const int*   batch      = (const int*)d_in[3];
  const float* visit_node = (const float*)d_in[4];
  const float* ehr_nodes  = (const float*)d_in[5];
  const float* node_emb_w = (const float*)d_in[6];
  const float* edge_emb_w = (const float*)d_in[7];
  const float* lin_w      = (const float*)d_in[8];
  const float* lin_b      = (const float*)d_in[9];
  const float* alpha_w    = (const float*)d_in[10];
  const float* alpha_b    = (const float*)d_in[11];
  const float* beta_w     = (const float*)d_in[12];
  const float* beta_b     = (const float*)d_in[13];
  const float* wr_w       = (const float*)d_in[14];
  const float* wr_b       = (const float*)d_in[15];
  const float* conv_w     = (const float*)d_in[16];
  const float* conv_b     = (const float*)d_in[17];
  const float* mlp_w      = (const float*)d_in[18];
  const float* mlp_b      = (const float*)d_in[19];

  char* ws = (char*)d_ws;
  size_t off = 0;
  auto carve = [&](size_t bytes) -> void* {
    void* p = ws + off;
    off += (bytes + 255) & ~(size_t)255;
    return p;
  };
  // f32 scratch
  float* proj_node = (float*)carve((size_t)NN1 * H * 4);
  float* ubuf      = (float*)carve((size_t)L * D * 4);
  float* wrel      = (float*)carve((size_t)L * NE1 * 4);
  float* xA        = (float*)carve((size_t)N * H * 4);
  float* xB        = (float*)carve((size_t)N * H * 4);
  float* agg       = (float*)carve((size_t)N * H * 4);
  float* hbuf      = (float*)carve((size_t)B * V * NN1 * 4);
  float* betab     = (float*)carve((size_t)B * V * 4);
  float* attnb     = (float*)carve((size_t)B * NN1 * 4);
  float* xg        = (float*)carve((size_t)B * H * 4);
  float* denomb    = (float*)carve((size_t)B * 4);
  float* tbuf      = (float*)carve((size_t)B * D * 4);
  float* xn        = (float*)carve((size_t)B * H * 4);
  // bf16 padded GEMM operands
  hbf* a_emb   = (hbf*)carve((size_t)MP_EMB * D * 2);        // node_emb  [2048, 768]
  hbf* w_lin   = (hbf*)carve((size_t)H * D * 2);             // lin_w     [256, 768]
  hbf* a_vis   = (hbf*)carve((size_t)MP_VIS * KP_NN * 2);    // visit     [1280, 2016]
  hbf* w_alpha = (hbf*)carve((size_t)L * NP_NN * KP_NN * 2); // alpha_w   [2, 2048, 2016]
  hbf* a_conv  = (hbf*)carve((size_t)N * H * 2);             // agg + x   [32768, 256]
  hbf* w_conv  = (hbf*)carve((size_t)L * H * H * 2);         // conv_w    [2, 256, 256]

  const dim3 blk(256);
  auto cvt = [&](const float* s, const float* s2, hbf* dst, int R, int K, int Rp, int Kp) {
    int total = Rp * Kp;
    if (s2)
      cvt_bf16_kernel<true><<<(total + 255) / 256, 256, 0, stream>>>(s, s2, dst, R, K, Kp, total);
    else
      cvt_bf16_kernel<false><<<(total + 255) / 256, 256, 0, stream>>>(s, s, dst, R, K, Kp, total);
  };

  // ---- one-time operand staging (bf16, padded) ----
  cvt(node_emb_w, nullptr, a_emb, NN1, D, MP_EMB, D);
  cvt(lin_w, nullptr, w_lin, H, D, H, D);
  cvt(visit_node, nullptr, a_vis, B * V, NN1, MP_VIS, KP_NN);
  for (int l = 0; l < L; ++l) {
    cvt(alpha_w + (size_t)l * NN1 * NN1, nullptr, w_alpha + (size_t)l * NP_NN * KP_NN,
        NN1, NN1, NP_NN, KP_NN);
    cvt(conv_w + (size_t)l * H * H, nullptr, w_conv + (size_t)l * H * H, H, H, H, H);
  }

  // proj_node = node_emb_w @ lin_w^T + lin_b   [2001, 256]
  {
    dim3 g(MP_EMB / 128, H / 64);
    wmma_gemm_bf16<false><<<g, blk, 0, stream>>>(a_emb, w_lin, lin_b, proj_node, NN1, H, D);
  }
  // per-edge-id relation scalar table (kills the [E,768]x[768,256] GEMM)
  fold_wr_kernel<<<(L * D + 255) / 256, 256, 0, stream>>>(wr_w, lin_w, ubuf);
  wrel_kernel<<<(L * NE1 + 255) / 256, 256, 0, stream>>>(edge_emb_w, ubuf, lin_b, wr_b ? wr_w : wr_w, wr_b, wrel);
  // x0 = proj_node[node_ids]
  gather_rows_kernel<<<N, 64, 0, stream>>>(node_ids, proj_node, xA);

  float* xin = xA;
  float* xout = xB;
  for (int l = 0; l < L; ++l) {
    beta_kernel<<<(B * V + 255) / 256, 256, 0, stream>>>(
        visit_node, beta_w + (size_t)l * NN1, beta_b + l, betab);
    {  // h = visit_node @ alpha_w[l]^T + alpha_b[l]   [1280, 2001] — dominant GEMM
      dim3 g(MP_VIS / 128, NP_NN / 64);
      wmma_gemm_bf16<false><<<g, blk, 0, stream>>>(
          a_vis, w_alpha + (size_t)l * NP_NN * KP_NN, alpha_b + (size_t)l * NN1,
          hbuf, B * V, NN1, KP_NN);
    }
    softmax_attn_kernel<<<(B * NN1 + 255) / 256, 256, 0, stream>>>(hbuf, betab, attnb);
    zero_kernel<<<((N * H / 4) + 255) / 256, 256, 0, stream>>>((float4*)agg, N * H / 4);
    edge_msg_kernel<<<E, 64, 0, stream>>>(
        edge_index, edge_ids, node_ids, batch, attnb, wrel + (size_t)l * NE1, xin, agg);
    // x' = relu((agg + x) @ conv_w[l]^T + conv_b[l])   [32768, 256]
    cvt(agg, xin, a_conv, N, H, N, H);
    {
      dim3 g(N / 128, H / 64);
      wmma_gemm_bf16<true><<<g, blk, 0, stream>>>(
          a_conv, w_conv + (size_t)l * H * H, conv_b + (size_t)l * H, xout, N, H, H);
    }
    float* tmp = xin; xin = xout; xout = tmp;
  }

  pool_kernel<<<B, H, 0, stream>>>(xin, xg);
  denom_kernel<<<1, 64, 0, stream>>>(ehr_nodes, denomb);
  tnode_kernel<<<(B * D + 255) / 256, 256, 0, stream>>>(ehr_nodes, node_emb_w, denomb, tbuf);
  xnode_kernel<<<(B * H + 255) / 256, 256, 0, stream>>>(tbuf, lin_w, lin_b, xn);
  mlp_kernel<<<(B * OUT + 255) / 256, 256, 0, stream>>>(xg, xn, mlp_w, mlp_b, (float*)d_out);
}
